// CausalAttention_54597624266924
// MI455X (gfx1250) — matrix operand order repaired and checked by reading, not compiled, not run on hardware
//
#include <hip/hip_runtime.h>
#include <hip/hip_bf16.h>

#define S_LEN  2048
#define BATCH  2
#define HID    2048
#define NH     32
#define NKV    8
#define HD     64
#define M_TOT  (BATCH * S_LEN)   // 4096
#define QN     (NH * HD)         // 2048
#define KVN    (NKV * HD)        // 512

typedef __attribute__((ext_vector_type(16))) __bf16         v16bf;
typedef __attribute__((ext_vector_type(8)))  float          v8f;
typedef __attribute__((ext_vector_type(8)))  unsigned short v8u16;
typedef __attribute__((ext_vector_type(4)))  unsigned int   v4u;
typedef __attribute__((ext_vector_type(8)))  int            v8i;
typedef __attribute__((ext_vector_type(4)))  int            v4i;

union Frag16 {
  v16bf bf;
  struct { v8u16 lo, hi; } h;
};

__device__ __forceinline__ unsigned short f2bf(float f) {
  unsigned int u = __float_as_uint(f);
  u += 0x7FFFu + ((u >> 16) & 1u);       // round-to-nearest-even
  return (unsigned short)(u >> 16);
}

// ---------------------------------------------------------------- TDM 2D tile load (D# per CDNA5 ISA)
// tile_w elems contiguous (dim0), tile_h rows, row_stride elems, bf16 (2B).
// LDS padding: pad_amount_code+1 DWORDs inserted every 2<<pad_interval_code DWORDs.
__device__ __forceinline__ void tdm_load_2d(unsigned lds_off, const void* gptr,
                                            unsigned tile_w, unsigned tile_h,
                                            unsigned row_stride,
                                            unsigned pad_iv, unsigned pad_am) {
  unsigned long long ga = (unsigned long long)(size_t)gptr;
  v4u g0;
  g0.x = 1u;                                       // count=1, user mode
  g0.y = lds_off;                                  // lds_addr (bytes)
  g0.z = (unsigned)(ga & 0xFFFFFFFFu);             // global_addr[31:0]
  g0.w = (unsigned)((ga >> 32) & 0x01FFFFFFu) | (2u << 30);  // addr[56:32] | type=2
  v8i g1;
  g1[0] = (int)((1u << 16) | (1u << 20) | (pad_iv << 22) | (pad_am << 25)); // data_size=2B, pad_enable
  g1[1] = (int)((tile_w & 0xFFFFu) << 16);         // tensor_dim0[15:0] == tile_w (no OOB)
  g1[2] = (int)((tile_h & 0xFFFFu) << 16);         // tensor_dim1[15:0] == tile_h
  g1[3] = (int)((tile_w & 0xFFFFu) << 16);         // tile_dim0
  g1[4] = (int)(tile_h & 0xFFFFu);                 // tile_dim1 (tile_dim2 = 0)
  g1[5] = (int)row_stride;                         // tensor_dim0_stride[31:0]
  g1[6] = 0;
  g1[7] = 0;
  v4i z4 = {0, 0, 0, 0};
  v8i z8 = {0, 0, 0, 0, 0, 0, 0, 0};
  __builtin_amdgcn_tensor_load_to_lds(g0, g1, z4, z4, z8, 0);  // 6-arg (clang-23)
}

// ---------------------------------------------------------------- convert
__global__ void k_f32_to_bf16(const float* __restrict__ in,
                              unsigned short* __restrict__ out, int n) {
  int i = blockIdx.x * blockDim.x + threadIdx.x;
  int stride = gridDim.x * blockDim.x;
  for (; i < n; i += stride) out[i] = f2bf(in[i]);
}

// ---------------------------------------------------------------- GEMM: C[m][n] = sum_k A[m][k] * W[n][k]
// One wave computes a 32x64 tile (2 row-groups x 4 col-groups), K-steps of 32.
// Optional fused RoPE epilogue (N tile width 64 == head_dim, head aligned).
__global__ void __launch_bounds__(32)
k_gemm_wt(const unsigned short* __restrict__ A,
          const unsigned short* __restrict__ W,
          unsigned short* __restrict__ outBf,
          float* __restrict__ outF,
          const float* __restrict__ cosT,
          const float* __restrict__ sinT,
          int M, int N, int K, int applyRope) {
  const int lane = threadIdx.x;
  const int hf   = lane >> 4;
  const int l16  = lane & 15;
  const int ntiles = N >> 6;
  const int tm = blockIdx.x / ntiles;
  const int tn = blockIdx.x % ntiles;
  const int m0 = tm << 5;
  const int n0 = tn << 6;

  v8f acc[2][4] = {{{}, {}, {}, {}}, {{}, {}, {}, {}}};

  const unsigned short* Arow0 = A + (size_t)(m0 + l16) * K;
  const unsigned short* Arow1 = A + (size_t)(m0 + 16 + l16) * K;
  for (int kt = 0; kt < K; kt += 32) {
    Frag16 fa0, fa1;
    fa0.h.lo = *(const v8u16*)(Arow0 + kt + hf * 8);
    fa0.h.hi = *(const v8u16*)(Arow0 + kt + 16 + hf * 8);
    fa1.h.lo = *(const v8u16*)(Arow1 + kt + hf * 8);
    fa1.h.hi = *(const v8u16*)(Arow1 + kt + 16 + hf * 8);
    if (kt + 32 < K) {
      __builtin_prefetch(Arow0 + kt + 32, 0, 3);
      __builtin_prefetch(Arow1 + kt + 32, 0, 3);
    }
#pragma unroll
    for (int j = 0; j < 4; ++j) {
      const unsigned short* Wrow =
          W + (size_t)(n0 + j * 16 + l16) * K + kt;
      Frag16 fb;
      fb.h.lo = *(const v8u16*)(Wrow + hf * 8);
      fb.h.hi = *(const v8u16*)(Wrow + 16 + hf * 8);
      acc[0][j] = __builtin_amdgcn_wmma_f32_16x16x32_bf16(
          false, fa0.bf, false, fb.bf, (short)0, acc[0][j], false, false);
      acc[1][j] = __builtin_amdgcn_wmma_f32_16x16x32_bf16(
          false, fa1.bf, false, fb.bf, (short)0, acc[1][j], false, false);
    }
  }

#pragma unroll
  for (int rg = 0; rg < 2; ++rg) {
    if (applyRope) {
      v8f outv[4] = {{}, {}, {}, {}};
#pragma unroll
      for (int j = 0; j < 4; ++j) {
        const int d = j * 16 + l16;
#pragma unroll
        for (int v = 0; v < 8; ++v) {
          const int m = m0 + rg * 16 + v + 8 * hf;     // m == b*S + s
          const float c  = cosT[(size_t)m * HD + d];
          const float sn = sinT[(size_t)m * HD + d];
          const float x   = acc[rg][j][v];
          const float rot = (j < 2) ? -acc[rg][j + 2][v] : acc[rg][j - 2][v];
          outv[j][v] = x * c + rot * sn;
        }
      }
#pragma unroll
      for (int j = 0; j < 4; ++j) acc[rg][j] = outv[j];
    }
#pragma unroll
    for (int j = 0; j < 4; ++j) {
#pragma unroll
      for (int v = 0; v < 8; ++v) {
        const int m = m0 + rg * 16 + v + 8 * hf;
        const size_t off = (size_t)m * N + n0 + j * 16 + l16;
        if (outF) outF[off] = acc[rg][j][v];
        else      outBf[off] = f2bf(acc[rg][j][v]);
      }
    }
  }
}

// ---------------------------------------------------------------- V transpose: (b,s,hkv,d) -> (b,hkv,d,s)
__global__ void k_transpose_v(const unsigned short* __restrict__ V,
                              unsigned short* __restrict__ VT) {
  int t = blockIdx.x * blockDim.x + threadIdx.x;
  const int total = BATCH * NKV * HD * S_LEN;
  if (t >= total) return;
  const int s   = t & (S_LEN - 1);
  const int d   = (t >> 11) & (HD - 1);
  const int hkv = (t >> 17) & (NKV - 1);
  const int b   = t >> 20;
  VT[t] = V[(size_t)(b * S_LEN + s) * KVN + hkv * HD + d];
}

// ---------------------------------------------------------------- flash attention (causal, GQA)
// One wave per 16-row q tile; 32-key blocks; K/V tiles streamed by the
// Tensor Data Mover into double-buffered LDS, synchronized via TENSORcnt.
#define KROW 72   // K LDS row pitch in elems (128B data + 16B pad = 144B)
#define VROW 40   // V LDS row pitch in elems (64B data + 16B pad = 80B)

__global__ void __launch_bounds__(32)
k_flash_attn(const unsigned short* __restrict__ Q,   // [b*S+s][h*64+d]
             const unsigned short* __restrict__ Kc,  // [b*S+k][hkv*64+d]
             const unsigned short* __restrict__ VT,  // [(b*NKV+hkv)*64+d][s]
             unsigned short* __restrict__ Y) {       // [b*S+s][h*64+d]
  const int lane = threadIdx.x, hf = lane >> 4, l16 = lane & 15;
  const int qtile = blockIdx.x % (S_LEN / 16);
  const int h     = (blockIdx.x / (S_LEN / 16)) % NH;
  const int b     = blockIdx.x / ((S_LEN / 16) * NH);
  const int hkv   = h / (NH / NKV);
  const int m0    = qtile * 16;

  __shared__ __align__(16) unsigned short Klds[2][32 * KROW];
  __shared__ __align__(16) unsigned short Vlds[2][64 * VROW];
  __shared__ __align__(16) unsigned short Plds[16 * 40];

  const unsigned koff[2] = {(unsigned)(size_t)&Klds[0][0],
                            (unsigned)(size_t)&Klds[1][0]};
  const unsigned voff[2] = {(unsigned)(size_t)&Vlds[0][0],
                            (unsigned)(size_t)&Vlds[1][0]};

  const unsigned short* Kbase = Kc + (size_t)(b * S_LEN) * KVN + hkv * HD;
  const unsigned short* Vbase = VT + (size_t)((b * NKV + hkv) * HD) * S_LEN;

  // Q A-fragments, two 32-wide d chunks
  Frag16 qa[2];
  const unsigned short* qrow = Q + (size_t)(b * S_LEN + m0 + l16) * QN + h * HD;
#pragma unroll
  for (int c = 0; c < 2; ++c) {
    qa[c].h.lo = *(const v8u16*)(qrow + c * 32 + hf * 8);
    qa[c].h.hi = *(const v8u16*)(qrow + c * 32 + 16 + hf * 8);
  }

  v8f acco[4] = {{}, {}, {}, {}};
  float mrun[8], lrun[8];
#pragma unroll
  for (int v = 0; v < 8; ++v) { mrun[v] = -1e30f; lrun[v] = 0.f; }

  const int nkb = (m0 + 16 + 31) >> 5;

  // TDM: prefetch block 0 (K tile 32x64 @stride 512; V^T tile 64x32 @stride 2048)
  tdm_load_2d(koff[0], Kbase, 64, 32, KVN, 4, 3);     // pad: every 32 DW, +4 DW
  tdm_load_2d(voff[0], Vbase, 32, 64, S_LEN, 3, 3);   // pad: every 16 DW, +4 DW

  for (int kb = 0; kb < nkb; ++kb) {
    const int k0 = kb * 32;
    const int buf = kb & 1;
    if (kb + 1 < nkb) {   // issue next block's tiles, then retire current pair
      tdm_load_2d(koff[buf ^ 1], Kbase + (size_t)(k0 + 32) * KVN, 64, 32, KVN, 4, 3);
      tdm_load_2d(voff[buf ^ 1], Vbase + (k0 + 32), 32, 64, S_LEN, 3, 3);
      __builtin_amdgcn_s_wait_tensorcnt(2);
    } else {
      __builtin_amdgcn_s_wait_tensorcnt(0);
    }

    // ---- scores: two 16x16 tiles (key groups g), K-dim = 64 = 2 chunks
    v8f sc[2];
#pragma unroll
    for (int g = 0; g < 2; ++g) {
      const unsigned short* kl = &Klds[buf][(g * 16 + l16) * KROW];
      v8f t = {};
#pragma unroll
      for (int c = 0; c < 2; ++c) {
        Frag16 fk;
        fk.h.lo = *(const v8u16*)(kl + c * 32 + hf * 8);
        fk.h.hi = *(const v8u16*)(kl + c * 32 + 16 + hf * 8);
        t = __builtin_amdgcn_wmma_f32_16x16x32_bf16(
            false, qa[c].bf, false, fk.bf, (short)0, t, false, false);
      }
      sc[g] = t;
    }
    // ---- scale + causal mask + row max (rows: m0 + v + 8*hf)
    float rmax[8];
#pragma unroll
    for (int v = 0; v < 8; ++v) {
      const int row = m0 + v + 8 * hf;
#pragma unroll
      for (int g = 0; g < 2; ++g) {
        float s = sc[g][v] * 0.125f;
        if (k0 + g * 16 + l16 > row) s = -1e30f;
        sc[g][v] = s;
      }
      float t = fmaxf(sc[0][v], sc[1][v]);
      t = fmaxf(t, __shfl_xor(t, 1, 32));
      t = fmaxf(t, __shfl_xor(t, 2, 32));
      t = fmaxf(t, __shfl_xor(t, 4, 32));
      t = fmaxf(t, __shfl_xor(t, 8, 32));
      rmax[v] = t;
    }
    // ---- online softmax update
#pragma unroll
    for (int v = 0; v < 8; ++v) {
      const float mnew  = fmaxf(mrun[v], rmax[v]);
      const float alpha = __expf(mrun[v] - mnew);
      float rs = 0.f;
#pragma unroll
      for (int g = 0; g < 2; ++g) {
        const float p = __expf(sc[g][v] - mnew);
        sc[g][v] = p;
        rs += p;
      }
      rs += __shfl_xor(rs, 1, 32);
      rs += __shfl_xor(rs, 2, 32);
      rs += __shfl_xor(rs, 4, 32);
      rs += __shfl_xor(rs, 8, 32);
      lrun[v] = lrun[v] * alpha + rs;
      mrun[v] = mnew;
#pragma unroll
      for (int j = 0; j < 4; ++j) acco[j][v] *= alpha;
    }
    // ---- P: C-layout -> LDS -> A-fragment layout
#pragma unroll
    for (int g = 0; g < 2; ++g)
#pragma unroll
      for (int v = 0; v < 8; ++v)
        Plds[(v + 8 * hf) * 40 + g * 16 + l16] = f2bf(sc[g][v]);
    __syncthreads();
    Frag16 pa;
    pa.h.lo = *(const v8u16*)(&Plds[l16 * 40 + hf * 8]);
    pa.h.hi = *(const v8u16*)(&Plds[l16 * 40 + 16 + hf * 8]);
    __syncthreads();
    // ---- O += P * V  (B fragment rows from padded V LDS tile)
#pragma unroll
    for (int j = 0; j < 4; ++j) {
      const unsigned short* vl = &Vlds[buf][(j * 16 + l16) * VROW];
      Frag16 fv;
      fv.h.lo = *(const v8u16*)(vl + hf * 8);
      fv.h.hi = *(const v8u16*)(vl + 16 + hf * 8);
      acco[j] = __builtin_amdgcn_wmma_f32_16x16x32_bf16(
          false, pa.bf, false, fv.bf, (short)0, acco[j], false, false);
    }
  }

  // ---- normalize + store Y (bf16, rows of 2048 = [b,s,h,d])
#pragma unroll
  for (int v = 0; v < 8; ++v) {
    const float inv = 1.0f / lrun[v];
    const int m = b * S_LEN + m0 + v + 8 * hf;
#pragma unroll
    for (int j = 0; j < 4; ++j)
      Y[(size_t)m * QN + h * HD + j * 16 + l16] = f2bf(acco[j][v] * inv);
  }
}

// ---------------------------------------------------------------- launch
extern "C" void kernel_launch(void* const* d_in, const int* in_sizes, int n_in,
                              void* d_out, int out_size, void* d_ws, size_t ws_size,
                              hipStream_t stream) {
  (void)in_sizes; (void)n_in; (void)out_size; (void)ws_size;
  const float* X    = (const float*)d_in[0];
  const float* cosT = (const float*)d_in[1];
  const float* sinT = (const float*)d_in[2];
  const float* Wq   = (const float*)d_in[3];
  const float* Wk   = (const float*)d_in[4];
  const float* Wv   = (const float*)d_in[5];
  const float* Wo   = (const float*)d_in[6];
  float* out = (float*)d_out;

  unsigned short* p = (unsigned short*)d_ws;
  unsigned short* Xbf = p;  p += (size_t)M_TOT * HID;
  unsigned short* Wqb = p;  p += (size_t)QN * HID;
  unsigned short* Wkb = p;  p += (size_t)KVN * HID;
  unsigned short* Wvb = p;  p += (size_t)KVN * HID;
  unsigned short* Wob = p;  p += (size_t)HID * HID;
  unsigned short* Qb  = p;  p += (size_t)M_TOT * QN;
  unsigned short* Kb  = p;  p += (size_t)M_TOT * KVN;
  unsigned short* Vb  = p;  p += (size_t)M_TOT * KVN;
  unsigned short* VTb = p;  p += (size_t)M_TOT * KVN;
  unsigned short* Yb  = p;

  auto conv = [&](const float* s, unsigned short* d, int n) {
    int blocks = (n + 255) / 256; if (blocks > 32768) blocks = 32768;
    k_f32_to_bf16<<<blocks, 256, 0, stream>>>(s, d, n);
  };
  conv(X,  Xbf, M_TOT * HID);
  conv(Wq, Wqb, QN * HID);
  conv(Wk, Wkb, KVN * HID);
  conv(Wv, Wvb, KVN * HID);
  conv(Wo, Wob, HID * HID);

  k_gemm_wt<<<(M_TOT / 32) * (QN / 64), 32, 0, stream>>>(
      Xbf, Wqb, Qb, nullptr, cosT, sinT, M_TOT, QN, HID, 1);
  k_gemm_wt<<<(M_TOT / 32) * (KVN / 64), 32, 0, stream>>>(
      Xbf, Wkb, Kb, nullptr, cosT, sinT, M_TOT, KVN, HID, 1);
  k_gemm_wt<<<(M_TOT / 32) * (KVN / 64), 32, 0, stream>>>(
      Xbf, Wvb, Vb, nullptr, nullptr, nullptr, M_TOT, KVN, HID, 0);

  k_transpose_v<<<(BATCH * NKV * HD * S_LEN) / 256, 256, 0, stream>>>(Vb, VTb);

  k_flash_attn<<<BATCH * NH * (S_LEN / 16), 32, 0, stream>>>(Qb, Kb, VTb, Yb);

  k_gemm_wt<<<(M_TOT / 32) * (HID / 64), 32, 0, stream>>>(
      Yb, Wob, nullptr, out, nullptr, nullptr, M_TOT, HID, HID, 0);
}
